// NeuralMemory_43628277793427
// MI455X (gfx1250) — matrix operand order repaired and checked by reading, not compiled, not run on hardware
//
// CDNA5 / gfx1250 NeuralMemory (Titans-style) forward.
//
// Key transformation: grad_fn always uses the BASE weights, so the scan
//   s_{i+1} = DECAY*s_i - LR*g_i   collapses to   s = -LR * sum_i DECAY^(nc-1-i) g_i
// => all 128 chunk gradients are independent; the weighted sums become two
// single GEMMs with K=32768 by folding c_i = -LR*0.9^(127-i) into the B
// operand at load time (deterministic, no atomics).
//
// Precision: split-fp32 -> (hi,lo) bf16 pairs; every tile does 3x
// v_wmma_f32_16x16x32_bf16 (hi*hi + hi*lo + lo*hi), ~fp32-quality results at
// ~2.7x the throughput of the f32 WMMA (16x16x4) path.
//
// Workspace requirement: ~680 MB (see layout in kernel_launch).

#include <hip/hip_runtime.h>

typedef __attribute__((ext_vector_type(16))) __bf16 v16bf;
typedef __attribute__((ext_vector_type(8)))  __bf16 v8bf;
typedef __attribute__((ext_vector_type(8)))  float  v8f;

#define BM 64
#define BN 64
#define BK 32
#define LDSS 40   // 32 + 8 pad (bf16 elems): 80B row stride, 16B aligned, conflict-free

enum { EPI_STORE = 0, EPI_HGELU = 1, EPI_SUBV = 2, EPI_DGELU = 3, EPI_GELU = 5 };

__device__ __forceinline__ float gelu_f(float x) {
  return 0.5f * x * (1.0f + erff(x * 0.7071067811865475f));
}
__device__ __forceinline__ float dgelu_f(float x) {
  return 0.5f * (1.0f + erff(x * 0.7071067811865475f))
       + x * 0.3989422804014327f * expf(-0.5f * x * x);
}
__device__ __forceinline__ void store_split(__bf16* Hh, __bf16* Hl, int idx, float v) {
  __bf16 h = (__bf16)v;
  Hh[idx] = h;
  Hl[idx] = (__bf16)(v - (float)h);
}

#define SHUF16(a, b) __builtin_shufflevector(a, b, 0,1,2,3,4,5,6,7,8,9,10,11,12,13,14,15)

// Generic split-bf16 WMMA GEMM:  C = epi( A(MxK) * B(KxN) ), batched over z.
//  TRANS_A: A stored K-major (storage row-major KxM, lda = row length)
//  TRANS_B: B stored N-major (storage row-major NxK, ldb = row length)
//  SCALE_B: multiply B elements by -LR * DECAY^(nchunks-1 - k/256) at load
//  PERM_OUT: remap output row (b*8192 + i*64 + r) -> (i*256 + b*64 + r)
template <bool TRANS_A, bool TRANS_B, int EPI, bool PERM_OUT, bool SCALE_B>
__global__ __launch_bounds__(256) void gemm_wmma(
    const float* __restrict__ A, int lda, long long aOffZ,
    const float* __restrict__ B, int ldb,
    float* __restrict__ C, int ldc, long long cOffZ,
    float* __restrict__ AUX, long long auxOffZ,
    int M, int N, int Kt, float alpha)
{
  __shared__ __attribute__((aligned(16))) __bf16 Ah[BM * LDSS];
  __shared__ __attribute__((aligned(16))) __bf16 Al[BM * LDSS];
  __shared__ __attribute__((aligned(16))) __bf16 Bh[BN * LDSS];
  __shared__ __attribute__((aligned(16))) __bf16 Bl[BN * LDSS];

  const int t  = threadIdx.x;
  const int m0 = blockIdx.y * BM;
  const int n0 = blockIdx.x * BN;
  const long long z = blockIdx.z;
  A += z * aOffZ;
  C += z * cOffZ;
  AUX += (AUX ? z * auxOffZ : 0);

  const int wave = t >> 5, lane = t & 31;
  const int hlf = lane >> 4, l15 = lane & 15;
  const int wm = wave & 3;   // 16-row block within 64
  const int wn = wave >> 2;  // 32-col block within 64 (two 16x16 tiles)

  v8f acc[2] = {};
  const int nch_m1 = (Kt >> 8) - 1;  // only meaningful when SCALE_B

  for (int k0 = 0; k0 < Kt; k0 += BK) {
    float cb = 1.0f;
    if (SCALE_B) cb = -0.01f * __powf(0.9f, (float)(nch_m1 - (k0 >> 8)));

    // ---- stage A tile (64 x 32 fp32 -> hi/lo bf16 in LDS, [m][k]) ----
    if (!TRANS_A) {
      const int mt = t >> 2, kb = (t & 3) << 3;
      const float* src = A + (long long)(m0 + mt) * lda + (k0 + kb);
#pragma unroll
      for (int u = 0; u < 8; ++u)
        store_split(Ah, Al, mt * LDSS + kb + u, src[u]);
    } else {
      const int kt = t >> 3, mb = (t & 7) << 3;
      const float* src = A + (long long)(k0 + kt) * lda + (m0 + mb);
#pragma unroll
      for (int u = 0; u < 8; ++u)
        store_split(Ah, Al, (mb + u) * LDSS + kt, src[u]);
    }
    // ---- stage B tile (32 x 64 fp32 -> hi/lo bf16 in LDS, [n][k]) ----
    if (!TRANS_B) {
      const int kt = t >> 3, nb = (t & 7) << 3;
      const float* src = B + (long long)(k0 + kt) * ldb + (n0 + nb);
#pragma unroll
      for (int u = 0; u < 8; ++u) {
        float v = src[u];
        if (SCALE_B) v *= cb;
        store_split(Bh, Bl, (nb + u) * LDSS + kt, v);
      }
    } else {
      const int nt = t >> 2, kb = (t & 3) << 3;
      const float* src = B + (long long)(n0 + nt) * ldb + (k0 + kb);
#pragma unroll
      for (int u = 0; u < 8; ++u) {
        float v = src[u];
        if (SCALE_B) v *= cb;
        store_split(Bh, Bl, nt * LDSS + kb + u, v);
      }
    }
    __syncthreads();

    // ---- fragments (ISA 16-bit A 16x32 / B 32x16 layouts) ----
    const int aoff = (wm * 16 + l15) * LDSS + hlf * 8;
    const v16bf aHv = SHUF16(*(const v8bf*)&Ah[aoff], *(const v8bf*)&Ah[aoff + 16]);
    const v16bf aLv = SHUF16(*(const v8bf*)&Al[aoff], *(const v8bf*)&Al[aoff + 16]);
#pragma unroll
    for (int tile = 0; tile < 2; ++tile) {
      const int boff = (wn * 32 + tile * 16 + l15) * LDSS + hlf * 8;
      const v16bf bHv = SHUF16(*(const v8bf*)&Bh[boff], *(const v8bf*)&Bh[boff + 16]);
      const v16bf bLv = SHUF16(*(const v8bf*)&Bl[boff], *(const v8bf*)&Bl[boff + 16]);
      acc[tile] = __builtin_amdgcn_wmma_f32_16x16x32_bf16(
          false, aHv, false, bHv, (short)0, acc[tile], false, false);
      acc[tile] = __builtin_amdgcn_wmma_f32_16x16x32_bf16(
          false, aHv, false, bLv, (short)0, acc[tile], false, false);
      acc[tile] = __builtin_amdgcn_wmma_f32_16x16x32_bf16(
          false, aLv, false, bHv, (short)0, acc[tile], false, false);
    }
    __syncthreads();
  }

  // ---- epilogue (C layout: VGPR r -> row r + hlf*8, col = lane&15) ----
#pragma unroll
  for (int tile = 0; tile < 2; ++tile) {
    const int gcol = n0 + wn * 32 + tile * 16 + l15;
#pragma unroll
    for (int r = 0; r < 8; ++r) {
      const int grow = m0 + wm * 16 + hlf * 8 + r;
      const float v = acc[tile][r];
      long long orow = grow;
      if (PERM_OUT) {  // (b,i,r) -> chunk-major
        const int bb = grow >> 13, rem = grow & 8191;
        orow = (long long)(((rem >> 6) << 8) + (bb << 6) + (rem & 63));
      }
      const long long idx = orow * (long long)ldc + gcol;
      if (EPI == EPI_STORE)       C[idx] = alpha * v;
      else if (EPI == EPI_HGELU)  { AUX[idx] = v; C[idx] = gelu_f(v); }
      else if (EPI == EPI_SUBV)   C[idx] = alpha * (v - AUX[idx]);
      else if (EPI == EPI_DGELU)  C[idx] = v * dgelu_f(AUX[idx]);   // in-place safe
      else                        C[idx] = gelu_f(v);
    }
  }
}

__global__ void fw_combine(float* __restrict__ dst, const float* __restrict__ w,
                           const float* __restrict__ s, int n) {
  const int i = blockIdx.x * blockDim.x + threadIdx.x;
  if (i < n) dst[i] = 0.9f * w[i] + s[i];
}

extern "C" void kernel_launch(void* const* d_in, const int* in_sizes, int n_in,
                              void* d_out, int out_size, void* d_ws, size_t ws_size,
                              hipStream_t stream) {
  (void)in_sizes; (void)n_in; (void)out_size; (void)ws_size;
  const float* x  = (const float*)d_in[0];
  const float* wk = (const float*)d_in[1];
  const float* wv = (const float*)d_in[2];
  const float* wq = (const float*)d_in[3];
  const float* W0 = (const float*)d_in[4];
  const float* W1 = (const float*)d_in[5];
  float* out = (float*)d_out;

  const int D = 512, H = 1024, R = 32768;      // R = B*S rows
  const long long RD = (long long)R * D, RH = (long long)R * H;

  float* ws = (float*)d_ws;
  size_t o = 0;
  float* Kc = ws + o; o += RD;   // k, chunk-major rows
  float* Vc = ws + o; o += RD;   // v, chunk-major rows
  float* Qb = ws + o; o += RD;   // q, natural rows
  float* Hb = ws + o; o += RH;   // pre-gelu h, later overwritten by dh
  float* Ab = ws + o; o += RH;   // a = gelu(h)
  float* Eb = ws + o; o += RD;   // e = (a@W1 - v) * 2/numel
  float* S0 = ws + o; o += (size_t)D * H;
  float* S1 = ws + o; o += (size_t)H * D;
  float* f0 = ws + o; o += (size_t)D * H;
  float* f1 = ws + o; o += (size_t)H * D;
  float* hq = ws + o; o += RH;   // gelu(q @ fw0)

  const dim3 blk(256);
  const float escale = 2.0f / 131072.0f;  // 2 / numel of (4,64,512)

  // 1) projections: K,V chunk-major; Q natural
  gemm_wmma<false,false,EPI_STORE,true ,false><<<dim3(8, 512, 1), blk, 0, stream>>>(
      x, D, 0, wk, D, Kc, D, 0, nullptr, 0, R, D, D, 1.0f);
  gemm_wmma<false,false,EPI_STORE,true ,false><<<dim3(8, 512, 1), blk, 0, stream>>>(
      x, D, 0, wv, D, Vc, D, 0, nullptr, 0, R, D, D, 1.0f);
  gemm_wmma<false,false,EPI_STORE,false,false><<<dim3(8, 512, 1), blk, 0, stream>>>(
      x, D, 0, wq, D, Qb, D, 0, nullptr, 0, R, D, D, 1.0f);

  // 2) per-chunk forward (z = 128 chunks of 256 rows): h = k@W0 ; a = gelu(h)
  gemm_wmma<false,false,EPI_HGELU,false,false><<<dim3(16, 4, 128), blk, 0, stream>>>(
      Kc, D, 256LL * D, W0, H, Ab, H, 256LL * H, Hb, 256LL * H, 256, H, D, 1.0f);
  // e = (a@W1 - v) * 2/numel
  gemm_wmma<false,false,EPI_SUBV,false,false><<<dim3(8, 4, 128), blk, 0, stream>>>(
      Ab, H, 256LL * H, W1, D, Eb, D, 256LL * D, Vc, 256LL * D, 256, D, H, escale);
  // S1 = sum_i c_i * a_iT @ e_i  ->  single GEMM, K=32768, c folded into e
  gemm_wmma<true ,false,EPI_STORE,false,true ><<<dim3(8, 16, 1), blk, 0, stream>>>(
      Ab, H, 0, Eb, D, S1, D, 0, nullptr, 0, H, D, R, 1.0f);
  // dh = (e @ W1^T) * gelu'(h)   (overwrites Hb in place)
  gemm_wmma<false,true ,EPI_DGELU,false,false><<<dim3(16, 4, 128), blk, 0, stream>>>(
      Eb, D, 256LL * D, W1, D, Hb, H, 256LL * H, Hb, 256LL * H, 256, H, D, 1.0f);
  // S0 = sum_i c_i * k_iT @ dh_i  ->  single GEMM, K=32768
  gemm_wmma<true ,false,EPI_STORE,false,true ><<<dim3(16, 8, 1), blk, 0, stream>>>(
      Kc, D, 0, Hb, H, S0, H, 0, nullptr, 0, D, H, R, 1.0f);

  // 3) fast weights: fw = (1-ALPHA)*W + S
  fw_combine<<<dim3((D * H + 255) / 256), blk, 0, stream>>>(f0, W0, S0, D * H);
  fw_combine<<<dim3((H * D + 255) / 256), blk, 0, stream>>>(f1, W1, S1, H * D);

  // 4) retrieval: out = gelu(Q @ fw0) @ fw1
  gemm_wmma<false,false,EPI_GELU ,false,false><<<dim3(16, 512, 1), blk, 0, stream>>>(
      Qb, D, 0, f0, H, hq, H, 0, nullptr, 0, R, H, D, 1.0f);
  gemm_wmma<false,false,EPI_STORE,false,false><<<dim3(8, 512, 1), blk, 0, stream>>>(
      hq, H, 0, f1, D, out, D, 0, nullptr, 0, R, D, H, 1.0f);
}
